// GroupedQueryAttention_58720792871822
// MI455X (gfx1250) — compile-verified, matrix operand order checked by reading, not run on hardware
//
#include <hip/hip_runtime.h>
#include <hip/hip_bf16.h>
#include <math.h>

// ---------------------------------------------------------------------------
// GQA forward for MI455X (gfx1250, wave32, WMMA + TDM).
// All matmuls: v_wmma_f32_16x16x32_bf16. A-tiles DMA'd via tensor_load_to_lds
// (double-buffered, s_wait_tensorcnt) when the builtin exists.
// Workspace requirement: 88,080,384 bytes.
// ---------------------------------------------------------------------------

typedef __attribute__((ext_vector_type(16))) __bf16 v16bf;
typedef __attribute__((ext_vector_type(8)))  __bf16 v8bf;
typedef __attribute__((ext_vector_type(4)))  __bf16 v4bf;
typedef __attribute__((ext_vector_type(8)))  float  v8f;
typedef __attribute__((ext_vector_type(4)))  float  f32x4;
typedef __attribute__((ext_vector_type(4)))  unsigned int u32x4;
typedef __attribute__((ext_vector_type(8)))  int    i32x8;
typedef __attribute__((ext_vector_type(4)))  int    i32x4;

#define HIDDEN   2048
#define NHEADS   16
#define NKV      4
#define HEADDIM  128
#define SEQ      2048
#define BATCH    2
#define ROWS     (BATCH * SEQ)
#define QK_SCALE 0.08838834764831845f   // 128^-0.5

#if defined(__gfx1250__) && __has_builtin(__builtin_amdgcn_tensor_load_to_lds) && \
    __has_builtin(__builtin_amdgcn_s_wait_tensorcnt)
#define USE_TDM 1
#else
#define USE_TDM 0
#endif

__device__ __forceinline__ v8f wmma_bf16(v16bf a, v16bf b, v8f c) {
    return __builtin_amdgcn_wmma_f32_16x16x32_bf16(
        false, a, false, b, (short)0, c, false, false);
}

// One lane's 16x32 A/B fragment from a row of >=32 bf16 (ISA 7.12.2 layout:
// lane<16 -> K in {0..7} u {16..23}, lane>=16 -> K in {8..15} u {24..31}).
__device__ __forceinline__ v16bf frag16(const __bf16* p, int half) {
    const v8bf lo = *(const v8bf*)(p + half * 8);
    const v8bf hi = *(const v8bf*)(p + half * 8 + 16);
    v16bf f;
#pragma unroll
    for (int i = 0; i < 8; ++i) { f[i] = lo[i]; f[i + 8] = hi[i]; }
    return f;
}

// ---------------------------------------------------------------------------
// f32 -> bf16 bulk conversion (one-shot, bandwidth-trivial)
// ---------------------------------------------------------------------------
__global__ __launch_bounds__(256)
void tobf16(const float* __restrict__ in, __bf16* __restrict__ out, int n4) {
    int i = blockIdx.x * 256 + threadIdx.x;
    if (i >= n4) return;
    f32x4 v = ((const f32x4*)in)[i];
    v4bf o = {(__bf16)v[0], (__bf16)v[1], (__bf16)v[2], (__bf16)v[3]};
    ((v4bf*)out)[i] = o;
}

// ---------------------------------------------------------------------------
// GEMM: C[M,N] = A[M,K] @ B[K,N], bf16 in global, f32 accumulate.
// Block 128x128, K-step 32, double-buffered LDS. A-tile via TDM (padded to the
// 48-elem LDS stride), B-tile via global b128 -> transposed LDS stores.
// ---------------------------------------------------------------------------
#define TM 128
#define TN 128
#define TK 32
#define LDSS 48   // padded bf16 row stride (96 B; 16 B-aligned chunks)

template <bool OUTBF16>
__global__ __launch_bounds__(256)
void gemm_bf16(const __bf16* __restrict__ A, const __bf16* __restrict__ B,
               void* __restrict__ Cv, int M, int N, int K) {
    __shared__ __align__(16) __bf16 sA[2][TM][LDSS];
    __shared__ __align__(16) __bf16 sB[2][TN][LDSS];   // [n][k]

    const int tid  = threadIdx.x;
    const int lane = tid & 31, wave = tid >> 5;
    const int half = lane >> 4, l16 = lane & 15;
    const int wm = wave >> 1, wn = wave & 1;           // 4 x 2 waves
    const int bm = blockIdx.y * TM, bn = blockIdx.x * TN;

    const int ar  = tid >> 1, ac = (tid & 1) * 16;     // A fallback loader
    const int bk  = tid >> 4, bn0 = (tid & 15) * 8;    // B loader

    v8f acc[2][4];
#pragma unroll
    for (int i = 0; i < 2; ++i)
#pragma unroll
        for (int j = 0; j < 4; ++j) acc[i][j] = {};

#if USE_TDM
    // 2-D TDM descriptor: tile 32(k) x 128(m) of 2B elems, LDS-padded
    // 8 DWORDs after every 16 DWORDs -> 48-element rows matching LDSS.
    auto issue_tdm = [&](int k0, int buf) {
        if (wave != 0) return;
        unsigned lds_off = (unsigned)(uintptr_t)&sA[buf][0][0];
        unsigned long long ga =
            (unsigned long long)(uintptr_t)(A + (size_t)bm * K + k0);
        u32x4 g0;
        g0[0] = 1u;                                          // count=1
        g0[1] = lds_off;                                     // lds_addr
        g0[2] = (unsigned)(ga & 0xffffffffu);                // global_addr lo
        g0[3] = (unsigned)((ga >> 32) & 0x01ffffffu) | (2u << 30); // hi | type=2
        i32x8 g1;
        g1[0] = (1 << 16) | (1 << 20) | (3 << 22) | (7 << 25); // 2B, pad 16+8 DW
        g1[1] = (int)(((unsigned)K & 0xffffu) << 16);          // dim0 lo16
        g1[2] = (int)(((unsigned)K >> 16) | (((unsigned)M & 0xffffu) << 16));
        g1[3] = (int)(((unsigned)M >> 16) | (32u << 16));      // tile_dim0=32
        g1[4] = 128;                                           // tile_dim1=128
        g1[5] = (int)K;                                        // dim0 stride
        g1[6] = 0; g1[7] = 0;
        i32x4 gz4 = {0, 0, 0, 0};                              // groups 2/3: NULL
        i32x8 gz8 = {0, 0, 0, 0, 0, 0, 0, 0};                  // unused trailing group
        __builtin_amdgcn_tensor_load_to_lds(g0, g1, gz4, gz4, gz8, 0);
    };
#endif
    auto load_b = [&](int k0, v8bf* rb) {
        rb[0] = *(const v8bf*)&B[(size_t)(k0 + bk) * N + bn + bn0];
        rb[1] = *(const v8bf*)&B[(size_t)(k0 + bk + 16) * N + bn + bn0];
    };
    auto store_b = [&](int buf, const v8bf* rb) {
#pragma unroll
        for (int i = 0; i < 8; ++i) sB[buf][bn0 + i][bk]      = rb[0][i];
#pragma unroll
        for (int i = 0; i < 8; ++i) sB[buf][bn0 + i][bk + 16] = rb[1][i];
    };
    auto load_a = [&](int k0, v8bf* ra) {
        const __bf16* p = &A[(size_t)(bm + ar) * K + k0 + ac];
        ra[0] = *(const v8bf*)p;
        ra[1] = *(const v8bf*)(p + 8);
    };
    auto store_a = [&](int buf, const v8bf* ra) {
        *(v8bf*)&sA[buf][ar][ac]     = ra[0];
        *(v8bf*)&sA[buf][ar][ac + 8] = ra[1];
    };

    // ---- prologue: tile 0 into buffer 0 ----
    {
        v8bf rb0[2];
        load_b(0, rb0);
        store_b(0, rb0);
#if USE_TDM
        issue_tdm(0, 0);
        if (wave == 0) __builtin_amdgcn_s_wait_tensorcnt(0);
#else
        v8bf ra0[2];
        load_a(0, ra0);
        store_a(0, ra0);
#endif
    }
    __syncthreads();

    const int nk = K / TK;
    int cur = 0;
    for (int kt = 0; kt < nk; ++kt) {
        const bool more = (kt + 1) < nk;
        v8bf rbn[2];
#if USE_TDM
        if (more) { issue_tdm((kt + 1) * TK, cur ^ 1); load_b((kt + 1) * TK, rbn); }
#else
        v8bf ran[2];
        if (more) { load_a((kt + 1) * TK, ran); load_b((kt + 1) * TK, rbn); }
#endif
        v16bf af[2], bfr[4];
#pragma unroll
        for (int i = 0; i < 2; ++i)
            af[i] = frag16(&sA[cur][wm * 32 + i * 16 + l16][0], half);
#pragma unroll
        for (int j = 0; j < 4; ++j)
            bfr[j] = frag16(&sB[cur][wn * 64 + j * 16 + l16][0], half);
#pragma unroll
        for (int i = 0; i < 2; ++i)
#pragma unroll
            for (int j = 0; j < 4; ++j)
                acc[i][j] = wmma_bf16(af[i], bfr[j], acc[i][j]);

        if (more) {
            store_b(cur ^ 1, rbn);
#if USE_TDM
            if (wave == 0) __builtin_amdgcn_s_wait_tensorcnt(0);
#else
            store_a(cur ^ 1, ran);
#endif
        }
        __syncthreads();
        cur ^= 1;
    }

#pragma unroll
    for (int i = 0; i < 2; ++i)
#pragma unroll
        for (int j = 0; j < 4; ++j)
#pragma unroll
            for (int r = 0; r < 8; ++r) {
                const int row = bm + wm * 32 + i * 16 + r + half * 8;
                const int col = bn + wn * 64 + j * 16 + l16;
                const float v = acc[i][j][r];
                if constexpr (OUTBF16)
                    ((__bf16*)Cv)[(size_t)row * N + col] = (__bf16)v;
                else
                    ((float*)Cv)[(size_t)row * N + col] = v;
            }
}

// ---------------------------------------------------------------------------
// RMS-norm + RoPE + repack (projections arrive bf16).
//   head 0..15 -> Q (scale folded) [B,H,S,D];  16..19 -> K [B,KV,S,D];
//   20..23 -> V transposed [B,KV,D,S].
// ---------------------------------------------------------------------------
__global__ __launch_bounds__(128)
void norm_rope_pack(const __bf16* __restrict__ Qp, const __bf16* __restrict__ Kp,
                    const __bf16* __restrict__ Vp, const float* __restrict__ cosT,
                    const float* __restrict__ sinT, const float* __restrict__ qw,
                    const float* __restrict__ kw, __bf16* __restrict__ Qbf,
                    __bf16* __restrict__ Kbf, __bf16* __restrict__ Vt) {
    const int row  = blockIdx.x;     // b*SEQ + s
    const int head = blockIdx.y;     // 0..23
    const int d    = threadIdx.x;
    const int b    = row / SEQ, s = row % SEQ;

    float x;
    if (head < NHEADS)
        x = (float)Qp[(size_t)row * (NHEADS * HEADDIM) + head * HEADDIM + d];
    else if (head < NHEADS + NKV)
        x = (float)Kp[(size_t)row * (NKV * HEADDIM) + (head - NHEADS) * HEADDIM + d];
    else
        x = (float)Vp[(size_t)row * (NKV * HEADDIM) + (head - NHEADS - NKV) * HEADDIM + d];

    if (head >= NHEADS + NKV) {
        int kv = head - NHEADS - NKV;
        Vt[((size_t)(b * NKV + kv) * HEADDIM + d) * SEQ + s] = (__bf16)x;
        return;
    }

    __shared__ float sh[HEADDIM];
    __shared__ float ssum[4];
    float sq = x * x;
#pragma unroll
    for (int m = 1; m < 32; m <<= 1) sq += __shfl_xor(sq, m, 32);
    if ((threadIdx.x & 31) == 0) ssum[threadIdx.x >> 5] = sq;
    __syncthreads();
    float var = (ssum[0] + ssum[1] + ssum[2] + ssum[3]) * (1.0f / HEADDIM);
    float inv = rsqrtf(var + 1e-6f);
    float xn  = x * inv * (head < NHEADS ? qw[d] : kw[d]);
    sh[d] = xn;
    __syncthreads();
    float rot = (d < HEADDIM / 2) ? -sh[d + HEADDIM / 2] : sh[d - HEADDIM / 2];
    float o   = xn * cosT[(size_t)s * HEADDIM + d] + rot * sinT[(size_t)s * HEADDIM + d];

    if (head < NHEADS) {
        o *= QK_SCALE;
        Qbf[((size_t)(b * NHEADS + head) * SEQ + s) * HEADDIM + d] = (__bf16)o;
    } else {
        Kbf[((size_t)(b * NKV + head - NHEADS) * SEQ + s) * HEADDIM + d] = (__bf16)o;
    }
}

// ---------------------------------------------------------------------------
// Flash attention, one wave32 per block, 16 query rows, causal, bf16 output.
// ---------------------------------------------------------------------------
__global__ __launch_bounds__(32)
void flash_attn(const __bf16* __restrict__ Qbf, const __bf16* __restrict__ Kbf,
                const __bf16* __restrict__ Vt, __bf16* __restrict__ Aout) {
    __shared__ __align__(16) __bf16 sP[16][32];

    const int qb   = blockIdx.x;
    const int bh   = blockIdx.y;
    const int b    = bh >> 4, h = bh & 15, kv = h >> 2;
    const int lane = threadIdx.x, half = lane >> 4, l16 = lane & 15;
    const int q0   = qb * 16;

    const __bf16* qbase = Qbf + ((size_t)bh * SEQ + q0) * HEADDIM;
    const __bf16* kbase = Kbf + ((size_t)(b * NKV + kv) * SEQ) * HEADDIM;
    const __bf16* vbase = Vt  + ((size_t)(b * NKV + kv) * HEADDIM) * SEQ;

    v16bf aq[4];
#pragma unroll
    for (int j = 0; j < 4; ++j)
        aq[j] = frag16(qbase + (size_t)l16 * HEADDIM + 32 * j, half);

    v8f o[8];
#pragma unroll
    for (int j = 0; j < 8; ++j) o[j] = {};
    float mrun[8], lrun[8];
#pragma unroll
    for (int r = 0; r < 8; ++r) { mrun[r] = -__builtin_inff(); lrun[r] = 0.f; }

    const int nkb = qb / 2 + 1;
    for (int kb = 0; kb < nkb; ++kb) {
        const int kbeg = kb * 32;

        v8f s[2];
#pragma unroll
        for (int t = 0; t < 2; ++t) {
            s[t] = {};
            const __bf16* kptr = kbase + (size_t)(kbeg + 16 * t + l16) * HEADDIM;
#pragma unroll
            for (int j = 0; j < 4; ++j)
                s[t] = wmma_bf16(aq[j], frag16(kptr + 32 * j, half), s[t]);
        }

#pragma unroll
        for (int r = 0; r < 8; ++r) {
            const int rowg = q0 + r + half * 8;
#pragma unroll
            for (int t = 0; t < 2; ++t)
                if (kbeg + 16 * t + l16 > rowg) s[t][r] = -__builtin_inff();

            float mx = fmaxf(s[0][r], s[1][r]);
#pragma unroll
            for (int m = 1; m < 16; m <<= 1) mx = fmaxf(mx, __shfl_xor(mx, m, 32));
            const float mnew = fmaxf(mrun[r], mx);
            const float corr = __expf(mrun[r] - mnew);
            const float p0   = __expf(s[0][r] - mnew);
            const float p1   = __expf(s[1][r] - mnew);
            float rs = p0 + p1;
#pragma unroll
            for (int m = 1; m < 16; m <<= 1) rs += __shfl_xor(rs, m, 32);
            lrun[r] = lrun[r] * corr + rs;
            mrun[r] = mnew;
#pragma unroll
            for (int j = 0; j < 8; ++j) o[j][r] *= corr;

            sP[r + half * 8][l16]      = (__bf16)p0;
            sP[r + half * 8][l16 + 16] = (__bf16)p1;
        }
        // single wave per block: LDS in-order within a wave; drain DScnt
        asm volatile("s_wait_dscnt 0" ::: "memory");

        const v16bf pa = frag16(&sP[l16][0], half);
#pragma unroll
        for (int j = 0; j < 8; ++j) {
            const __bf16* vptr = vbase + (size_t)(16 * j + l16) * SEQ + kbeg;
            o[j] = wmma_bf16(pa, frag16(vptr, half), o[j]);
        }
    }

#pragma unroll
    for (int r = 0; r < 8; ++r) {
        const float inv = 1.0f / lrun[r];
        const int rowg = b * SEQ + q0 + r + half * 8;
#pragma unroll
        for (int j = 0; j < 8; ++j)
            Aout[(size_t)rowg * (NHEADS * HEADDIM) + h * HEADDIM + 16 * j + l16] =
                (__bf16)(o[j][r] * inv);
    }
}

// ---------------------------------------------------------------------------
extern "C" void kernel_launch(void* const* d_in, const int* in_sizes, int n_in,
                              void* d_out, int out_size, void* d_ws, size_t ws_size,
                              hipStream_t stream) {
    const float* X    = (const float*)d_in[0];
    const float* cosT = (const float*)d_in[1];
    const float* sinT = (const float*)d_in[2];
    const float* Wq   = (const float*)d_in[3];
    const float* Wk   = (const float*)d_in[4];
    const float* Wv   = (const float*)d_in[5];
    const float* Wo   = (const float*)d_in[6];
    const float* qw   = (const float*)d_in[7];
    const float* kw   = (const float*)d_in[8];

    char* ws = (char*)d_ws;
    __bf16* Xbf = (__bf16*)(ws);                 // 16,777,216
    __bf16* Wqb = (__bf16*)(ws + 16777216);      //  8,388,608
    __bf16* Wkb = (__bf16*)(ws + 25165824);      //  2,097,152
    __bf16* Wvb = (__bf16*)(ws + 27262976);      //  2,097,152
    __bf16* Wob = (__bf16*)(ws + 29360128);      //  8,388,608
    __bf16* Qpb = (__bf16*)(ws + 37748736);      // 16,777,216
    __bf16* Kpb = (__bf16*)(ws + 54525952);      //  4,194,304
    __bf16* Vpb = (__bf16*)(ws + 58720256);      //  4,194,304
    __bf16* Qbf = (__bf16*)(ws + 62914560);      // 16,777,216
    __bf16* Kbf = (__bf16*)(ws + 79691776);      //  4,194,304
    __bf16* Vt  = (__bf16*)(ws + 83886080);      //  4,194,304  (end 88,080,384)
    __bf16* Abf = Xbf;   // attention output reuses bf16-X buffer

    auto cvt = [&](const float* src, __bf16* dst, int n) {
        int n4 = n / 4;
        tobf16<<<(n4 + 255) / 256, 256, 0, stream>>>(src, dst, n4);
    };
    cvt(X,  Xbf, ROWS * HIDDEN);
    cvt(Wq, Wqb, HIDDEN * NHEADS * HEADDIM);
    cvt(Wk, Wkb, HIDDEN * NKV * HEADDIM);
    cvt(Wv, Wvb, HIDDEN * NKV * HEADDIM);
    cvt(Wo, Wob, NHEADS * HEADDIM * HIDDEN);

    gemm_bf16<true><<<dim3(HIDDEN / TN, ROWS / TM), 256, 0, stream>>>(
        Xbf, Wqb, Qpb, ROWS, NHEADS * HEADDIM, HIDDEN);
    gemm_bf16<true><<<dim3((NKV * HEADDIM) / TN, ROWS / TM), 256, 0, stream>>>(
        Xbf, Wkb, Kpb, ROWS, NKV * HEADDIM, HIDDEN);
    gemm_bf16<true><<<dim3((NKV * HEADDIM) / TN, ROWS / TM), 256, 0, stream>>>(
        Xbf, Wvb, Vpb, ROWS, NKV * HEADDIM, HIDDEN);

    norm_rope_pack<<<dim3(ROWS, NHEADS + 2 * NKV), 128, 0, stream>>>(
        Qpb, Kpb, Vpb, cosT, sinT, qw, kw, Qbf, Kbf, Vt);

    flash_attn<<<dim3(SEQ / 16, BATCH * NHEADS), 32, 0, stream>>>(Qbf, Kbf, Vt, Abf);

    gemm_bf16<false><<<dim3(HIDDEN / TN, ROWS / TM), 256, 0, stream>>>(
        Abf, Wob, (float*)d_out, ROWS, HIDDEN, NHEADS * HEADDIM);
}
